// MultiHeadedAttention_79688823210425
// MI455X (gfx1250) — matrix operand order repaired and checked by reading, not compiled, not run on hardware
//
#include <hip/hip_runtime.h>

// ---------------------------------------------------------------------------
// Types for CDNA5 WMMA (wave32) and TDM descriptors
// ---------------------------------------------------------------------------
typedef __attribute__((ext_vector_type(16))) __bf16 bf16x16;
typedef __attribute__((ext_vector_type(8)))  float  floatx8;
typedef unsigned int u32x4 __attribute__((ext_vector_type(4)));
typedef int          i32x8 __attribute__((ext_vector_type(8)));
typedef int          i32x4 __attribute__((ext_vector_type(4)));

union FragAB {
    bf16x16      v;
    unsigned int u[8];
};

__device__ __forceinline__ unsigned short f2bf(float f) {
    unsigned int u = __float_as_uint(f);
    u += 0x7FFFu + ((u >> 16) & 1u);        // round-to-nearest-even
    return (unsigned short)(u >> 16);
}
__device__ __forceinline__ float bf2f(unsigned short h) {
    return __uint_as_float(((unsigned int)h) << 16);
}

// ---------------------------------------------------------------------------
// Tensor Data Mover: 2-D tile load (bf16 elements) global -> LDS.
// D# layout per CDNA5 ISA.
// Issued once per workgroup (wave 0); EXEC is ignored by TDM.
// ---------------------------------------------------------------------------
__device__ __forceinline__ void tdm_load_2d_bf16(unsigned lds_off,
                                                 const void* gptr,
                                                 unsigned tile_x,      // elems
                                                 unsigned tile_y,      // rows
                                                 unsigned stride) {    // elems
    const unsigned long long ga = (unsigned long long)gptr;
    u32x4 g0;
    g0[0] = 1u;                                         // count=1, user mode
    g0[1] = lds_off;                                    // LDS byte address
    g0[2] = (unsigned)(ga & 0xFFFFFFFFu);               // global_addr[31:0]
    g0[3] = (unsigned)((ga >> 32) & 0x01FFFFFFu)        // global_addr[56:32]
          | (2u << 30);                                 // type = 2 ("image")
    const unsigned td0 = stride;                        // x extent for OOB
    const unsigned td1 = 1u << 20;                      // generous y extent
    i32x8 g1;
    g1[0] = (int)(1u << 16);                            // data_size=1 (2 bytes)
    g1[1] = (int)((td0 & 0xFFFFu) << 16);               // tensor_dim0 lo16
    g1[2] = (int)((td0 >> 16) & 0xFFFFu)                // tensor_dim0 hi16
          | (int)((td1 & 0xFFFFu) << 16);               // tensor_dim1 lo16
    g1[3] = (int)((td1 >> 16) & 0xFFFFu)                // tensor_dim1 hi16
          | (int)((tile_x & 0xFFFFu) << 16);            // tile_dim0
    g1[4] = (int)(tile_y & 0xFFFFu);                    // tile_dim1, tile_dim2=0
    g1[5] = (int)stride;                                // tensor_dim0_stride lo32
    g1[6] = 0;
    g1[7] = 0;
    const i32x4 gz4 = {0, 0, 0, 0};
    const i32x8 gz8 = {0, 0, 0, 0, 0, 0, 0, 0};
    __builtin_amdgcn_tensor_load_to_lds(g0, g1, gz4, gz4, gz8, 0);
}

// ---------------------------------------------------------------------------
// fp32 -> bf16 conversion (plain)
// ---------------------------------------------------------------------------
__global__ __launch_bounds__(256)
void cvt_bf16_kernel(const float* __restrict__ src, unsigned short* __restrict__ dst,
                     long long n) {
    long long i = (long long)blockIdx.x * blockDim.x + threadIdx.x;
    if (i < n) dst[i] = f2bf(src[i]);
}

// fp32 [K][N] -> bf16 [N][K] (weights transposed once so every GEMM B is [N][K])
__global__ __launch_bounds__(256)
void cvt_bf16_transpose_kernel(const float* __restrict__ src,
                               unsigned short* __restrict__ dst,
                               int Kdim, int Ndim) {
    const int idx = blockIdx.x * 256 + threadIdx.x;
    if (idx < Kdim * Ndim) {
        const int k = idx / Ndim, n = idx % Ndim;       // coalesced read
        dst[(long long)n * Kdim + k] = f2bf(src[idx]);
    }
}

// ---------------------------------------------------------------------------
// WMMA GEMM: C[M,N] = A[M,K] * B^T where B is row-major [N,K].
// Block tile 64x64x32, 4 waves (128 threads), 2x2 v_wmma_f32_16x16x32_bf16
// per wave. Both A and B tiles staged by the Tensor Data Mover into
// double-buffered LDS; wave 0 issues tile i+1 before waiting on tile i
// (s_wait_tensorcnt 2, in-order completion), overlapping DMA with WMMA.
// MODE 0: projection -> bf16 head-split [b,h,s,hd], (acc+bias)*scale
// MODE 1: scores     -> fp32, + qrel gather, mask -> -inf        (batched)
// MODE 2: context    -> fp32 out [z, m, n]                       (batched)
// MODE 3: output     -> fp32, + bias (final projection)
// MODE 4: V proj     -> bf16 head-split transposed [b,h,hd,s], acc+bias
// ---------------------------------------------------------------------------
#define BM 64
#define BN 64
#define BKK 32

template<int MODE>
__global__ __launch_bounds__(128)
void wmma_gemm_kernel(const unsigned short* __restrict__ Ag,
                      const unsigned short* __restrict__ Bg,
                      float* __restrict__ Cf,
                      unsigned short* __restrict__ Cbf,
                      const float* __restrict__ bias,
                      const float* __restrict__ qrel,
                      const unsigned char* __restrict__ mask,
                      int M, int N, int K, int lda, int ldb,
                      long long strideA, long long strideB, long long strideC,
                      float scale) {
    __shared__ unsigned short ldsA[2][BM][BKK];   // [buf][m][k]
    __shared__ unsigned short ldsB[2][BN][BKK];   // [buf][n][k]

    const int tid  = threadIdx.x;
    const int lane = tid & 31;
    const int wave = tid >> 5;
    const int z    = blockIdx.z;
    const int mBase = blockIdx.x * BM;
    const int nBase = blockIdx.y * BN;

    Ag += (long long)z * strideA + (long long)mBase * lda;
    Bg += (long long)z * strideB + (long long)nBase * ldb;

    const int waveM = (wave >> 1) * 32;
    const int waveN = (wave & 1) * 32;
    const int hi  = (lane >> 4) & 1;
    const int l15 = lane & 15;

    // LDS byte offsets for TDM descriptors (flat->LDS truncates to addr[31:0])
    const unsigned ldsA_off[2] = {
        (unsigned)(unsigned long long)(void*)&ldsA[0][0][0],
        (unsigned)(unsigned long long)(void*)&ldsA[1][0][0]};
    const unsigned ldsB_off[2] = {
        (unsigned)(unsigned long long)(void*)&ldsB[0][0][0],
        (unsigned)(unsigned long long)(void*)&ldsB[1][0][0]};

    floatx8 c[2][2];
#pragma unroll
    for (int i = 0; i < 2; ++i)
#pragma unroll
        for (int j = 0; j < 2; ++j)
#pragma unroll
            for (int r = 0; r < 8; ++r) c[i][j][r] = 0.0f;

    // prologue: stage tile 0 into buffer 0
    if (wave == 0) {
        tdm_load_2d_bf16(ldsA_off[0], Ag, BKK, BM, (unsigned)lda);
        tdm_load_2d_bf16(ldsB_off[0], Bg, BKK, BN, (unsigned)ldb);
    }

    for (int kt = 0; kt < K; kt += BKK) {
        const int buf = (kt / BKK) & 1;
        if (wave == 0) {
            if (kt + BKK < K) {
                // issue next tile into the other buffer, then wait for current
                tdm_load_2d_bf16(ldsA_off[buf ^ 1], Ag + kt + BKK, BKK, BM,
                                 (unsigned)lda);
                tdm_load_2d_bf16(ldsB_off[buf ^ 1], Bg + kt + BKK, BKK, BN,
                                 (unsigned)ldb);
                __builtin_amdgcn_s_wait_tensorcnt((short)2);
            } else {
                __builtin_amdgcn_s_wait_tensorcnt((short)0);
            }
        }
        __syncthreads();

        // ---- build fragments per ISA 16-bit A(16x32)/B(32x16) layouts ----
        FragAB a[2], b[2];
#pragma unroll
        for (int i = 0; i < 2; ++i) {
            const int m = waveM + i * 16 + l15;
#pragma unroll
            for (int j = 0; j < 8; ++j) {
                const int klo = (j < 4) ? (2 * j) : (16 + 2 * (j - 4));
                const int kk  = klo + hi * 8;    // lanes 16-31: +8
                a[i].u[j] = *reinterpret_cast<const unsigned int*>(&ldsA[buf][m][kk]);
            }
        }
#pragma unroll
        for (int j2 = 0; j2 < 2; ++j2) {
            const int n = waveN + j2 * 16 + l15;
#pragma unroll
            for (int j = 0; j < 8; ++j) {
                const int kk = ((j < 4) ? (2 * j) : (16 + 2 * (j - 4))) + hi * 8;  // lanes 16-31: +8
                b[j2].u[j] = *reinterpret_cast<const unsigned int*>(&ldsB[buf][n][kk]);
            }
        }

#pragma unroll
        for (int i = 0; i < 2; ++i)
#pragma unroll
            for (int j = 0; j < 2; ++j)
                c[i][j] = __builtin_amdgcn_wmma_f32_16x16x32_bf16(
                    false, a[i].v, false, b[j].v, (short)0, c[i][j], false, false);

        __syncthreads();   // releases this buffer for tile kt+2*BKK
    }

    // ---- epilogue: C element (vgpr r, lane) -> m = r + hi*8, n = lane&15 ----
#pragma unroll
    for (int i = 0; i < 2; ++i) {
#pragma unroll
        for (int j = 0; j < 2; ++j) {
#pragma unroll
            for (int r = 0; r < 8; ++r) {
                const int gm = mBase + waveM + i * 16 + r + hi * 8;
                const int gn = nBase + waveN + j * 16 + l15;
                float acc = c[i][j][r];
                if (MODE == 0) {
                    float v = (acc + bias[gn]) * scale;
                    const int bb = gm >> 10, s = gm & 1023;
                    const int h = gn >> 6, hd = gn & 63;
                    Cbf[(((long long)(bb * 16 + h)) * 1024 + s) * 64 + hd] = f2bf(v);
                } else if (MODE == 1) {
                    int d = gn - gm;
                    d = d < -32 ? -32 : (d > 32 ? 32 : d);
                    acc += qrel[((long long)z * 1024 + gm) * 65 + (d + 32)];
                    const int bb = z >> 4;
                    if (!mask[bb * 1024 + gn]) acc = -__builtin_inff();
                    Cf[(long long)z * strideC + (long long)gm * N + gn] = acc;
                } else if (MODE == 2) {
                    Cf[(long long)z * strideC + (long long)gm * N + gn] = acc;
                } else if (MODE == 3) {
                    Cf[(long long)gm * N + gn] = acc + bias[gn];
                } else { // MODE == 4: V projection, head-split transposed
                    float v = acc + bias[gn];
                    const int bb = gm >> 10, s = gm & 1023;
                    const int h = gn >> 6, hd = gn & 63;
                    Cbf[(((long long)(bb * 16 + h)) * 64 + hd) * 1024 + s] = f2bf(v);
                }
            }
        }
    }
}

// ---------------------------------------------------------------------------
// qrel[z,q,r] = q_head[z,q,:] . rel_k_emb[r,:]   (r = 0..64)
// ---------------------------------------------------------------------------
__global__ __launch_bounds__(96)
void qrel_kernel(const unsigned short* __restrict__ qh,
                 const float* __restrict__ relk,
                 float* __restrict__ qrel) {
    const long long row = blockIdx.x;            // z*1024 + q
    __shared__ float qrow[64];
    const int tid = threadIdx.x;
    if (tid < 64) qrow[tid] = bf2f(qh[row * 64 + tid]);
    __syncthreads();
    if (tid < 65) {
        float acc = 0.0f;
#pragma unroll
        for (int d = 0; d < 64; ++d) acc += qrow[d] * relk[tid * 64 + d];
        qrel[row * 65 + tid] = acc;
    }
}

// ---------------------------------------------------------------------------
// Softmax over k (length 1024) for all 16 heads of one (b,q);
// writes bf16 probs and the head-averaged attention directly into d_out.
// ---------------------------------------------------------------------------
__global__ __launch_bounds__(256)
void softmax_kernel(const float* __restrict__ scores,
                    unsigned short* __restrict__ probs,
                    float* __restrict__ attn_avg) {
    const int bq = blockIdx.x;                   // b*1024 + q
    const int b  = bq >> 10, q = bq & 1023;
    const int tid = threadIdx.x, lane = tid & 31, wv = tid >> 5;
    __shared__ float red[8];

    float avg[4] = {0.f, 0.f, 0.f, 0.f};
    for (int h = 0; h < 16; ++h) {
        const long long rowoff = (((long long)(b * 16 + h)) * 1024 + q) * 1024;
        const float* row = scores + rowoff;
        float x[4];
#pragma unroll
        for (int i = 0; i < 4; ++i) x[i] = row[tid + 256 * i];

        float m = fmaxf(fmaxf(x[0], x[1]), fmaxf(x[2], x[3]));
#pragma unroll
        for (int off = 16; off > 0; off >>= 1) m = fmaxf(m, __shfl_xor(m, off, 32));
        __syncthreads();
        if (lane == 0) red[wv] = m;
        __syncthreads();
        float mm = red[0];
#pragma unroll
        for (int w = 1; w < 8; ++w) mm = fmaxf(mm, red[w]);

        float p[4], s = 0.f;
#pragma unroll
        for (int i = 0; i < 4; ++i) { p[i] = __expf(x[i] - mm); s += p[i]; }
#pragma unroll
        for (int off = 16; off > 0; off >>= 1) s += __shfl_xor(s, off, 32);
        __syncthreads();
        if (lane == 0) red[wv] = s;
        __syncthreads();
        float ss = 0.f;
#pragma unroll
        for (int w = 0; w < 8; ++w) ss += red[w];
        const float inv = 1.0f / ss;

        unsigned short* prow = probs + rowoff;
#pragma unroll
        for (int i = 0; i < 4; ++i) {
            const float pv = p[i] * inv;
            prow[tid + 256 * i] = f2bf(pv);
            avg[i] += pv;
        }
    }
    float* arow = attn_avg + (long long)bq * 1024;
#pragma unroll
    for (int i = 0; i < 4; ++i) arow[tid + 256 * i] = avg[i] * (1.0f / 16.0f);
}

// ---------------------------------------------------------------------------
// rel-V bucket kernel: one block per (z,q).
// ---------------------------------------------------------------------------
__global__ __launch_bounds__(128)
void relv_kernel(const unsigned short* __restrict__ probs,
                 const float* __restrict__ relv,
                 const float* __restrict__ ctxf,
                 unsigned short* __restrict__ ctx_bf) {
    const int idx = blockIdx.x;                  // z*1024 + q
    const int z = idx >> 10, q = idx & 1023;
    const int b = z >> 4, h = z & 15;
    const int tid = threadIdx.x, lane = tid & 31, wv = tid >> 5;
    __shared__ float bucket[65];
    __shared__ float red[8];

    const unsigned short* prow = probs + (long long)idx * 1024;

    if (tid >= 1 && tid <= 63) {
        const int k = q + (tid - 32);
        bucket[tid] = (k >= 0 && k < 1024) ? bf2f(prow[k]) : 0.0f;
    }
    float s0 = 0.f, s64 = 0.f;
    for (int k = tid; k < 1024; k += 128) {
        const float pv = bf2f(prow[k]);
        if (k <= q - 32) s0 += pv;
        if (k >= q + 32) s64 += pv;
    }
#pragma unroll
    for (int off = 16; off > 0; off >>= 1) {
        s0  += __shfl_xor(s0, off, 32);
        s64 += __shfl_xor(s64, off, 32);
    }
    __syncthreads();
    if (lane == 0) { red[wv] = s0; red[4 + wv] = s64; }
    __syncthreads();
    if (tid == 0) {
        bucket[0]  = red[0] + red[1] + red[2] + red[3];
        bucket[64] = red[4] + red[5] + red[6] + red[7];
    }
    __syncthreads();

    if (tid < 64) {
        float acc = ctxf[(long long)idx * 64 + tid];
#pragma unroll 5
        for (int r = 0; r <= 64; ++r) acc += bucket[r] * relv[r * 64 + tid];
        ctx_bf[((long long)(b * 1024 + q)) * 1024 + h * 64 + tid] = f2bf(acc);
    }
}

// ---------------------------------------------------------------------------
// Host launcher
// ---------------------------------------------------------------------------
extern "C" void kernel_launch(void* const* d_in, const int* in_sizes, int n_in,
                              void* d_out, int out_size, void* d_ws, size_t ws_size,
                              hipStream_t stream) {
    (void)in_sizes; (void)n_in; (void)out_size; (void)ws_size;

    const float* key   = (const float*)d_in[0];
    const float* value = (const float*)d_in[1];
    const float* query = (const float*)d_in[2];
    const unsigned char* mask = (const unsigned char*)d_in[3];
    const float* Wq = (const float*)d_in[4];
    const float* bq = (const float*)d_in[5];
    const float* Wk = (const float*)d_in[6];
    const float* bk = (const float*)d_in[7];
    const float* Wv = (const float*)d_in[8];
    const float* bv = (const float*)d_in[9];
    const float* Wo = (const float*)d_in[10];
    const float* bo = (const float*)d_in[11];
    const float* relk = (const float*)d_in[12];
    const float* relv = (const float*)d_in[13];

    const long long ACT = 4LL * 1024 * 1024;     // [4096,1024]
    const long long WEL = 1024LL * 1024;

    char* ws = (char*)d_ws;
    auto alloc = [&](size_t bytes) -> char* {
        char* p = ws;
        ws += (bytes + 255) & ~(size_t)255;
        return p;
    };
    unsigned short* Xq  = (unsigned short*)alloc(ACT * 2);
    unsigned short* Xk  = (unsigned short*)alloc(ACT * 2);
    unsigned short* Xv  = (unsigned short*)alloc(ACT * 2);
    unsigned short* WqT = (unsigned short*)alloc(WEL * 2);   // [N][K] bf16
    unsigned short* WkT = (unsigned short*)alloc(WEL * 2);
    unsigned short* WvT = (unsigned short*)alloc(WEL * 2);
    unsigned short* WoT = (unsigned short*)alloc(WEL * 2);
    unsigned short* qh  = (unsigned short*)alloc(ACT * 2);   // [b,h,s,hd] bf16
    unsigned short* kh  = (unsigned short*)alloc(ACT * 2);   // [b,h,s,hd] bf16
    unsigned short* vhT = (unsigned short*)alloc(ACT * 2);   // [b,h,hd,s] bf16
    float* qrelbuf = (float*)alloc(65536LL * 65 * 4);        // [z,q,65]
    float* scores  = (float*)alloc(64LL * 1024 * 1024 * 4);  // [z,q,k] fp32
    unsigned short* probs = (unsigned short*)alloc(64LL * 1024 * 1024 * 2);
    float* ctxf    = (float*)alloc(ACT * 4);                 // [z,q,hd] fp32
    unsigned short* ctx_bf = (unsigned short*)alloc(ACT * 2);// [b,s,d] bf16

    float* output   = (float*)d_out;
    float* attn_avg = output + ACT;

    // 1) conversions to bf16 (weights transposed to [N][K])
    cvt_bf16_kernel<<<(unsigned)((ACT + 255) / 256), 256, 0, stream>>>(query, Xq, ACT);
    cvt_bf16_kernel<<<(unsigned)((ACT + 255) / 256), 256, 0, stream>>>(key,   Xk, ACT);
    cvt_bf16_kernel<<<(unsigned)((ACT + 255) / 256), 256, 0, stream>>>(value, Xv, ACT);
    cvt_bf16_transpose_kernel<<<(unsigned)((WEL + 255) / 256), 256, 0, stream>>>(Wq, WqT, 1024, 1024);
    cvt_bf16_transpose_kernel<<<(unsigned)((WEL + 255) / 256), 256, 0, stream>>>(Wk, WkT, 1024, 1024);
    cvt_bf16_transpose_kernel<<<(unsigned)((WEL + 255) / 256), 256, 0, stream>>>(Wv, WvT, 1024, 1024);
    cvt_bf16_transpose_kernel<<<(unsigned)((WEL + 255) / 256), 256, 0, stream>>>(Wo, WoT, 1024, 1024);

    // 2) Q/K/V projections (Q folds 1/sqrt(64); V written transposed)
    wmma_gemm_kernel<0><<<dim3(64, 16, 1), 128, 0, stream>>>(
        Xq, WqT, nullptr, qh, bq, nullptr, nullptr,
        4096, 1024, 1024, 1024, 1024, 0, 0, 0, 0.125f);
    wmma_gemm_kernel<0><<<dim3(64, 16, 1), 128, 0, stream>>>(
        Xk, WkT, nullptr, kh, bk, nullptr, nullptr,
        4096, 1024, 1024, 1024, 1024, 0, 0, 0, 1.0f);
    wmma_gemm_kernel<4><<<dim3(64, 16, 1), 128, 0, stream>>>(
        Xv, WvT, nullptr, vhT, bv, nullptr, nullptr,
        4096, 1024, 1024, 1024, 1024, 0, 0, 0, 1.0f);

    // 3) q . rel_k_emb (65 distance buckets per query row)
    qrel_kernel<<<65536, 96, 0, stream>>>(qh, relk, qrelbuf);

    // 4) scores = q @ k^T + rel gather, masked  (batched over z = b*H+h)
    wmma_gemm_kernel<1><<<dim3(16, 16, 64), 128, 0, stream>>>(
        qh, kh, scores, nullptr, nullptr, qrelbuf, mask,
        1024, 1024, 64, 64, 64, 65536, 65536, 1024LL * 1024, 1.0f);

    // 5) softmax + head-averaged attention into d_out
    softmax_kernel<<<4096, 256, 0, stream>>>(scores, probs, attn_avg);

    // 6) context = attn @ v  (batched; B = vhT is [hd][s] row-major)
    wmma_gemm_kernel<2><<<dim3(16, 1, 64), 128, 0, stream>>>(
        probs, vhT, ctxf, nullptr, nullptr, nullptr, nullptr,
        1024, 64, 1024, 1024, 1024, 1024LL * 1024, 65536, 65536, 1.0f);

    // 7) + rel-V term, merge heads to [b,s,d] bf16
    relv_kernel<<<65536, 128, 0, stream>>>(probs, relv, ctxf, ctx_bf);

    // 8) output = ctx @ Wo + bo  (fp32 into d_out)
    wmma_gemm_kernel<3><<<dim3(64, 16, 1), 128, 0, stream>>>(
        ctx_bf, WoT, output, nullptr, bo, nullptr, nullptr,
        4096, 1024, 1024, 1024, 1024, 0, 0, 0, 1.0f);
}
